// Hms2Model_58978490908797
// MI455X (gfx1250) — matrix operand order repaired and checked by reading, not compiled, not run on hardware
//
#include <hip/hip_runtime.h>
#include <hip/hip_bf16.h>

// ---------------------------------------------------------------------------
// HMS2 tiled conv pipeline == full-image two-conv pipeline:
//   img[1,2304,2304,3] int32 --/255--> conv1 3x3x3->64 s2 SAME relu -> a1[1152,1152,64] bf16
//   a1 -> conv2 3x3x64->128 s2 SAME relu -> out NCHW f32 [1,128,576,576]
// Both convs are implicit GEMMs on V_WMMA_F32_16X16X32_BF16.
// conv2 stages A-tile and B-chunks into LDS with the Tensor Data Mover
// (tensor_load_to_lds + s_wait_tensorcnt), double-buffered against WMMA.
// ---------------------------------------------------------------------------

typedef __attribute__((ext_vector_type(16))) __bf16 v16bf;
typedef __attribute__((ext_vector_type(8)))  float  v8f;
typedef __attribute__((ext_vector_type(4)))  unsigned int u32x4;
typedef __attribute__((ext_vector_type(8)))  int          i32x8;
typedef __attribute__((ext_vector_type(4)))  int          i32x4;

#if __has_builtin(__builtin_amdgcn_tensor_load_to_lds)
#define USE_TDM 1
#else
#define USE_TDM 0
#endif

union BFrag {
  v16bf v;
  unsigned short h[16];
  uint4 q[2];
};

__device__ __forceinline__ unsigned short f32_to_bf16(float f) {
  unsigned int u = __float_as_uint(f);
  u += 0x7FFFu + ((u >> 16) & 1u);          // round-to-nearest-even
  return (unsigned short)(u >> 16);
}

#define IMGW 2304
#define A1W  1152
#define A1C  64
#define OUTW 576
#define OUTC 128

#if USE_TDM
// ---------------------------------------------------------------------------
// Issue a TDM 2D tile load (elements are 2 bytes). D# packing per
// CDNA5 ISA:
//   group0: [1:0]=count=1, [63:32]=lds_addr, [120:64]=global_addr, [127:126]=type=2
//   group1: [17:16]=data_size(1=2B), [20]=pad_en, [24:22]=pad_interval,
//           [31:25]=pad_amount, [79:48]=tensor_dim0, [111:80]=tensor_dim1,
//           [127:112]=tile_dim0, [143:128]=tile_dim1, [207:160]=tensor_dim0_stride
// Dims/strides in data_size units. Completion tracked on TENSORcnt.
// ---------------------------------------------------------------------------
__device__ __forceinline__ void tdm_load_2d(unsigned long long gaddr, unsigned lds_addr,
                                            unsigned tile_d0, unsigned tile_d1,
                                            unsigned long long tensor_d0, unsigned tensor_d1,
                                            unsigned long long stride_d0,
                                            unsigned pad_en, unsigned pad_interval,
                                            unsigned pad_amount) {
  union { unsigned int u[4]; u32x4 v; } g0;
  union { int i[8]; i32x8 v; } g1;
  g0.u[0] = 1u;                                             // count=1 (valid), user mode
  g0.u[1] = lds_addr;                                       // LDS byte address
  g0.u[2] = (unsigned)(gaddr & 0xFFFFFFFFull);              // global_addr[31:0]
  g0.u[3] = (unsigned)((gaddr >> 32) & 0x01FFFFFFull) | (2u << 30);  // [56:32], type=2
  g1.i[0] = (int)((1u << 16) |                              // data_size = 2B
                  (pad_en ? ((1u << 20) | (pad_interval << 22) | (pad_amount << 25)) : 0u));
  g1.i[1] = (int)(((unsigned)(tensor_d0 & 0xFFFFull)) << 16);               // td0 lo16
  g1.i[2] = (int)(((unsigned)((tensor_d0 >> 16) & 0xFFFFull)) |             // td0 hi16
                  ((tensor_d1 & 0xFFFFu) << 16));                           // td1 lo16
  g1.i[3] = (int)(((tensor_d1 >> 16) & 0xFFFFu) | (tile_d0 << 16));         // td1 hi | tile0
  g1.i[4] = (int)(tile_d1 & 0xFFFFu);                                      // tile1, tile2=0
  g1.i[5] = (int)(unsigned)(stride_d0 & 0xFFFFFFFFull);                    // stride0 lo32
  g1.i[6] = (int)(unsigned)((stride_d0 >> 32) & 0xFFFFull);                // stride0 hi16
  g1.i[7] = 0;
  i32x4 z4 = (i32x4)0;
#if defined(__clang_major__) && (__clang_major__ >= 23)
  i32x8 z8 = (i32x8)0;
  __builtin_amdgcn_tensor_load_to_lds(g0.v, g1.v, z4, z4, z8, 0);
#else
  __builtin_amdgcn_tensor_load_to_lds(g0.v, g1.v, z4, z4, 0);
#endif
}
#endif // USE_TDM

// ---------------------------------------------------------------------------
// Weight pre-transpose into WMMA B-fragment order so each lane's 16 bf16
// values are contiguous (32B loads):
//   W1T[kh(2)][n(64)][kl(16)]         K = kl<8 ? kh*8+kl : 16+kh*8+kl-8 (27 real, pad->32)
//   W2T[t(18)][kh(2)][n(128)][kl(16)] Kg = t*32 + (kl<8 ? kh*8+kl : 16+kh*8+kl-8),  Kg=(ky*3+kx)*64+m
// ---------------------------------------------------------------------------
__global__ __launch_bounds__(256) void prep_weights(const float* __restrict__ W1,
                                                    const float* __restrict__ W2,
                                                    unsigned short* __restrict__ W1T,
                                                    unsigned short* __restrict__ W2T) {
  int idx = blockIdx.x * 256 + threadIdx.x;
  if (idx < 2048) {
    int kl = idx & 15, n = (idx >> 4) & 63, kh = idx >> 10;
    int K = (kl < 8) ? (kh * 8 + kl) : (16 + kh * 8 + (kl - 8));
    float v = (K < 27) ? W1[K * 64 + n] : 0.0f;   // W1 HWIO flat: K*64+n
    W1T[idx] = f32_to_bf16(v);
  }
  int j = idx - 2048;
  if (j >= 0 && j < 18 * 2 * 128 * 16) {
    int kl = j & 15, n = (j >> 4) & 127, kh = (j >> 11) & 1, t = j >> 12;
    int Kg = t * 32 + ((kl < 8) ? (kh * 8 + kl) : (16 + kh * 8 + (kl - 8)));
    W2T[j] = f32_to_bf16(W2[Kg * 128 + n]);       // W2 HWIO flat: Kg*128+n
  }
}

// ---------------------------------------------------------------------------
// conv1: block = 32 output pixels (one row) x 64 channels; 8 waves:
//   wave = (msub: which 16 pixels) x (nsub: which 16 channels); 1 WMMA each.
// ---------------------------------------------------------------------------
__global__ __launch_bounds__(256) void conv1_wmma(const int* __restrict__ img,
                                                  const float* __restrict__ b1,
                                                  const unsigned short* __restrict__ W1T,
                                                  unsigned short* __restrict__ a1) {
  __shared__ unsigned short simg[3][68][4];   // rows 2y..2y+2, cols 2x0..2x0+64, 3 ch
  const int y   = blockIdx.y;
  const int x0  = blockIdx.x * 32;
  const int tid = threadIdx.x;

  for (int idx = tid; idx < 3 * 65 * 3; idx += 256) {
    int r  = idx % 3;
    int c  = (idx / 3) % 65;
    int jy = idx / (3 * 65);
    int gy = 2 * y + jy, gx = 2 * x0 + c;
    float v = 0.0f;
    if (gy < IMGW && gx < IMGW)
      v = (float)img[(gy * IMGW + gx) * 3 + r] * (1.0f / 255.0f);
    simg[jy][c][r] = f32_to_bf16(v);
  }
  __syncthreads();

  const int lane = tid & 31;
  const int wv   = tid >> 5;
  const int msub = wv & 1;
  const int nsub = wv >> 1;
  const int ln   = lane & 15;
  const int hi   = lane >> 4;

  BFrag b;  // B 32x16: lanes 0-15 hold K {0..7,16..23}, lanes 16-31 hold K {8..15,24..31}
  {
    const uint4* p = (const uint4*)(W1T + ((hi * 64 + nsub * 16 + ln) << 4));
    b.q[0] = p[0];
    b.q[1] = p[1];
  }

  BFrag a;  // A 16x32: lanes 0-15 K {0..7,16..23}; lanes 16-31 K {8..15,24..31}
  const int pix = msub * 16 + ln;
#pragma unroll
  for (int s = 0; s < 16; ++s) {
    int v  = s >> 1, hh = s & 1;
    int K  = (v < 4) ? (hi * 8 + 2 * v + hh) : (16 + hi * 8 + 2 * (v - 4) + hh);
    unsigned short val = 0;
    if (K < 27) {                              // K = jy*9 + jx*3 + r
      int jy = K / 9, rem = K % 9;
      val = simg[jy][2 * pix + rem / 3][rem % 3];
    }
    a.h[s] = val;
  }

  v8f acc = {};
  acc = __builtin_amdgcn_wmma_f32_16x16x32_bf16(false, a.v, false, b.v,
                                                (short)0, acc, false, false);

  const int   ch   = nsub * 16 + ln;
  const float bias = b1[ch];
#pragma unroll
  for (int v = 0; v < 8; ++v) {
    int   M = v + hi * 8;                      // C: VGPR v -> M = v + 8*(lane>=16)
    int   x = x0 + msub * 16 + M;
    float f = acc[v] + bias;
    f = f > 0.0f ? f : 0.0f;
    a1[(y * A1W + x) * A1C + ch] = f32_to_bf16(f);
  }
}

// ---------------------------------------------------------------------------
// conv2: block = 8x8 output pixels x 128 channels. 8 waves:
//   wave = (mg: 16 of 64 pixels) x (nh: 64 of 128 channels -> 4 WMMA N-tiles).
// A tile (17x17x64 bf16 of a1, SA_PITCH-padded) staged once via TDM (interior
// blocks) or manual zero-filled loads (edge blocks). B 32x128 K-chunks (8KB)
// double-buffered via TDM, overlapped with WMMA; K=576, 18 steps of 32.
// ---------------------------------------------------------------------------
#define SA_PITCH 72                                   // pad 64->72 ushorts per (r,c)
#define SA_BYTES (17 * 17 * SA_PITCH * 2)             // 41616, 16B aligned

__global__ __launch_bounds__(256) void conv2_wmma(const unsigned short* __restrict__ a1,
                                                  const unsigned short* __restrict__ W2T,
                                                  const float* __restrict__ b2,
                                                  float* __restrict__ out) {
  __shared__ __align__(16) unsigned char smem[SA_BYTES + 2 * 8192];
  unsigned short* sa = (unsigned short*)smem;
  unsigned short* sb = (unsigned short*)(smem + SA_BYTES);

  const int tid = threadIdx.x;
  const int gy0 = blockIdx.y * 8, gx0 = blockIdx.x * 8;
  const int ar0 = 2 * gy0, ac0 = 2 * gx0;
  const bool interior = (blockIdx.y < (OUTW / 8) - 1) && (blockIdx.x < (OUTW / 8) - 1);

  const int lane = tid & 31;
  const int wv = tid >> 5;
  const int mg = wv & 3;
  const int nh = wv >> 2;
  const int ln = lane & 15;
  const int hi = lane >> 4;
  const int q  = mg * 16 + ln;                 // pixel within 8x8 block
  const int py = q >> 3, px = q & 7;

#if USE_TDM
  const unsigned lds_sa = (unsigned)(unsigned long long)(uintptr_t)sa;
  const unsigned lds_sb = (unsigned)(unsigned long long)(uintptr_t)sb;
#endif

  // ---- Stage A tile -------------------------------------------------------
#if USE_TDM
  if (!interior)
#endif
  {
    // Manual path (edge blocks, or no-TDM fallback): zero-fill = SAME padding.
    for (int idx = tid; idx < 17 * 17 * 8; idx += 256) {
      int ck = idx & 7;
      int cc = (idx >> 3) % 17;
      int rr = (idx >> 3) / 17;
      int r = ar0 + rr, c = ac0 + cc;
      uint4 val = make_uint4(0u, 0u, 0u, 0u);
      if (r < A1W && c < A1W)
        val = *(const uint4*)(a1 + ((r * A1W + c) * A1C + ck * 8));
      *(uint4*)(sa + (rr * 17 + cc) * SA_PITCH + ck * 8) = val;
    }
  }

#if USE_TDM
  if (wv == 0) {
    if (interior) {
      // 17 rows x 1088 contiguous elements (17 cols x 64 ch), row stride =
      // 1152*64 elems. LDS padding: 16B (pad_amount=3 -> 4 dwords) after every
      // 128B (pad_interval=4 -> 32 dwords) reproduces SA_PITCH=72 layout.
      tdm_load_2d((unsigned long long)(uintptr_t)(a1 + ((size_t)ar0 * A1W + ac0) * A1C),
                  lds_sa,
                  /*tile_d0=*/17 * 64, /*tile_d1=*/17,
                  /*tensor_d0=*/(unsigned long long)A1W * A1C, /*tensor_d1=*/A1W,
                  /*stride_d0=*/(unsigned long long)A1W * A1C,
                  /*pad_en=*/1, /*pad_interval=*/4, /*pad_amount=*/3);
    }
    // B chunk 0 (8KB contiguous) into buffer 0.
    tdm_load_2d((unsigned long long)(uintptr_t)W2T, lds_sb,
                4096, 1, 4096ull, 1, 4096ull, 0, 0, 0);
    __builtin_amdgcn_s_wait_tensorcnt(0);
  }
  __syncthreads();
#else
  // Fallback: synchronous copy of B chunk 0.
  __syncthreads();
  {
    const uint4* p = (const uint4*)W2T + tid * 2;
    uint4* d = (uint4*)sb;
    d[tid * 2] = p[0]; d[tid * 2 + 1] = p[1];
  }
  __syncthreads();
#endif

  v8f acc0 = {}, acc1 = {}, acc2 = {}, acc3 = {};

  for (int t = 0; t < 18; ++t) {
#if USE_TDM
    if (wv == 0 && t + 1 < 18)                 // async: DMA next chunk while computing
      tdm_load_2d((unsigned long long)(uintptr_t)(W2T + (t + 1) * 4096),
                  lds_sb + ((t + 1) & 1) * 8192,
                  4096, 1, 4096ull, 1, 4096ull, 0, 0, 0);
#endif
    const unsigned short* sbc = sb + (t & 1) * 4096;
    const int kk = t >> 1;                     // tap index: Kg = (ky*3+kx)*64 + m
    const int m0 = (t & 1) * 32;
    const int ky = kk / 3, kx = kk % 3;
    const int rr = 2 * py + ky, cc = 2 * px + kx;

    BFrag a;                                   // two contiguous b128s from LDS
    {
      const unsigned short* ap = sa + (rr * 17 + cc) * SA_PITCH + m0 + hi * 8;
      a.q[0] = *(const uint4*)ap;              // ch m0+{0..7 | 8..15}
      a.q[1] = *(const uint4*)(ap + 16);       // ch m0+{16..23 | 24..31}
    }
    BFrag b0, b1f, b2f, b3f;                   // 4 N-subtiles of 16, 32B each
    {
      const uint4* bp = (const uint4*)(sbc + ((hi * 128 + nh * 64 + ln) << 4));
      b0.q[0]  = bp[0];  b0.q[1]  = bp[1];
      b1f.q[0] = bp[32]; b1f.q[1] = bp[33];
      b2f.q[0] = bp[64]; b2f.q[1] = bp[65];
      b3f.q[0] = bp[96]; b3f.q[1] = bp[97];
    }
    acc0 = __builtin_amdgcn_wmma_f32_16x16x32_bf16(false, a.v, false, b0.v,  (short)0, acc0, false, false);
    acc1 = __builtin_amdgcn_wmma_f32_16x16x32_bf16(false, a.v, false, b1f.v, (short)0, acc1, false, false);
    acc2 = __builtin_amdgcn_wmma_f32_16x16x32_bf16(false, a.v, false, b2f.v, (short)0, acc2, false, false);
    acc3 = __builtin_amdgcn_wmma_f32_16x16x32_bf16(false, a.v, false, b3f.v, (short)0, acc3, false, false);

#if USE_TDM
    if (wv == 0 && t + 1 < 18)
      __builtin_amdgcn_s_wait_tensorcnt(0);    // issuing wave drains TDM
    __syncthreads();                           // releases all waves on next chunk
#else
    __syncthreads();
    if (t + 1 < 18) {
      const uint4* p = (const uint4*)(W2T + (t + 1) * 4096) + tid * 2;
      uint4 v0 = p[0], v1 = p[1];
      uint4* d = (uint4*)(sb + ((t + 1) & 1) * 4096);
      d[tid * 2] = v0; d[tid * 2 + 1] = v1;
      __syncthreads();
    }
#endif
  }

  // Epilogue: bias+relu, transpose via LDS (reuse sa region), coalesced NCHW stores
  __syncthreads();
  float* cbuf = (float*)smem;                  // [128][68] = 34816B < SA_BYTES
  {
    const int   chb = nh * 64 + ln;
    const float bs0 = b2[chb], bs1 = b2[chb + 16], bs2 = b2[chb + 32], bs3 = b2[chb + 48];
#pragma unroll
    for (int v = 0; v < 8; ++v) {
      int   qq = mg * 16 + v + hi * 8;
      float f;
      f = acc0[v] + bs0; cbuf[(chb)      * 68 + qq] = f > 0.0f ? f : 0.0f;
      f = acc1[v] + bs1; cbuf[(chb + 16) * 68 + qq] = f > 0.0f ? f : 0.0f;
      f = acc2[v] + bs2; cbuf[(chb + 32) * 68 + qq] = f > 0.0f ? f : 0.0f;
      f = acc3[v] + bs3; cbuf[(chb + 48) * 68 + qq] = f > 0.0f ? f : 0.0f;
    }
  }
  __syncthreads();
  for (int idx = tid; idx < 1024; idx += 256) {        // (ch, py) pairs
    int ch = idx >> 3, py2 = idx & 7;
    const float4* s = (const float4*)(cbuf + ch * 68 + py2 * 8);
    float4 lo = s[0], hi4 = s[1];
    float4* dst = (float4*)(out + ((size_t)ch * OUTW + (size_t)(gy0 + py2)) * OUTW + gx0);
    dst[0] = lo; dst[1] = hi4;
  }
}

// ---------------------------------------------------------------------------
extern "C" void kernel_launch(void* const* d_in, const int* in_sizes, int n_in,
                              void* d_out, int out_size, void* d_ws, size_t ws_size,
                              hipStream_t stream) {
  const int*   img = (const int*)d_in[0];
  const float* W1  = (const float*)d_in[1];
  const float* b1  = (const float*)d_in[2];
  const float* W2  = (const float*)d_in[3];
  const float* b2  = (const float*)d_in[4];
  float* out = (float*)d_out;

  // Workspace: a1 bf16 [1152*1152*64] (162MB) | W1T (4KB) | W2T (144KB)
  unsigned short* a1  = (unsigned short*)d_ws;
  unsigned short* W1T = a1 + (size_t)A1W * A1W * A1C;
  unsigned short* W2T = W1T + 2048;

  prep_weights<<<296, 256, 0, stream>>>(W1, W2, W1T, W2T);
  conv1_wmma<<<dim3(A1W / 32, A1W), 256, 0, stream>>>(img, b1, W1T, a1);
  conv2_wmma<<<dim3(OUTW / 8, OUTW / 8), 256, 0, stream>>>(a1, W2T, b2, out);
}
